// GNN_node_Virtualnode_79568564126476
// MI455X (gfx1250) — matrix operand order repaired and checked by reading, not compiled, not run on hardware
//
#include <hip/hip_runtime.h>
#include <hip/hip_bf16.h>
#include <stdint.h>

#define NNODES 50000
#define NEDGES 600000
#define NGRAPHS 512
#define EMB 128
#define INDIM 9
#define EDGEDIM 10
#define NLAYER 5
#define BN_EPS 1e-5f
#define NE (NNODES * EMB)          // 6,400,000
#define VE (NGRAPHS * EMB)         // 65,536

typedef __attribute__((ext_vector_type(16))) __bf16 bf16x16;
typedef __attribute__((ext_vector_type(8)))  float  f32x8;

union FragAB { bf16x16 v; unsigned u[8]; };

__device__ __forceinline__ unsigned short f2bf(float f) {
    unsigned u = __float_as_uint(f);
    u += 0x7FFFu + ((u >> 16) & 1u);       // round-to-nearest-even
    return (unsigned short)(u >> 16);
}

// ---------------------------------------------------------------------------
// BF16 WMMA GEMM: out[M,N] = A[M,K](bf16 row-major) * B[K,N] + bias
// B pre-packed FRAGMENT-MAJOR: u32 index ((ks*2+half)*N + n)*8 + v holds the
// bf16 pair (k, k+1) with kp = ks*16 + half*8 + v.  Each lane's B fragment is
// two contiguous uint4 loads; consecutive lanes are 32B apart (coalesced).
// A fragments: two contiguous uint4 loads per K-step, cached across N sweep.
// ---------------------------------------------------------------------------
template <int KSTEPS>
__global__ void gemm_bf16_wmma(const unsigned short* __restrict__ A,
                               const uint4* __restrict__ Bf,
                               const float* __restrict__ bias,
                               float* __restrict__ out,
                               int M, int N) {
    constexpr int K = KSTEPS * 32;
    const int wave = threadIdx.x >> 5;
    const int lane = threadIdx.x & 31;
    const int mt   = blockIdx.x * 4 + wave;        // 16-row tile index
    if (mt * 16 >= M) return;                       // wave-uniform -> EXEC stays full
    const int row  = mt * 16 + (lane & 15);
    const int half = lane >> 4;
    const int ncol = lane & 15;

    FragAB a[KSTEPS];
#pragma unroll
    for (int ks = 0; ks < KSTEPS; ++ks) {
        const uint4* ap = (const uint4*)(A + (size_t)row * K + ks * 32 + half * 16);
        const uint4 lo = ap[0];    // bf16 k = base+0..7   -> pairs (0,2,4,6)
        const uint4 hi = ap[1];    // bf16 k = base+8..15  -> pairs (8,10,12,14)
        a[ks].u[0] = lo.x; a[ks].u[1] = lo.y; a[ks].u[2] = lo.z; a[ks].u[3] = lo.w;
        a[ks].u[4] = hi.x; a[ks].u[5] = hi.y; a[ks].u[6] = hi.z; a[ks].u[7] = hi.w;
    }

    for (int n0 = 0; n0 < N; n0 += 16) {
        const float bv = bias[n0 + ncol];
        f32x8 acc = {bv, bv, bv, bv, bv, bv, bv, bv};
#pragma unroll
        for (int ks = 0; ks < KSTEPS; ++ks) {
            const size_t bi = ((size_t)((ks * 2 + half) * N + n0 + ncol)) * 2;
            const uint4 b0 = Bf[bi];
            const uint4 b1 = Bf[bi + 1];
            FragAB b;
            b.u[0] = b0.x; b.u[1] = b0.y; b.u[2] = b0.z; b.u[3] = b0.w;
            b.u[4] = b1.x; b.u[5] = b1.y; b.u[6] = b1.z; b.u[7] = b1.w;
            acc = __builtin_amdgcn_wmma_f32_16x16x32_bf16(
                false, a[ks].v, false, b.v, (short)0, acc, false, false);
        }
        const int rbase = mt * 16 + half * 8;
#pragma unroll
        for (int j = 0; j < 8; ++j)
            out[(size_t)(rbase + j) * N + n0 + ncol] = acc[j];
    }
}

// ---------------------------------------------------------------------------
// Weight packing: src[m][K][N] f32 -> fragment-major bf16-pair layout (above)
// ---------------------------------------------------------------------------
__global__ void pack_w(const float* __restrict__ src, unsigned* __restrict__ dst,
                       int K, int N, int nmat) {
    const size_t per   = (size_t)(K / 2) * N;
    const size_t total = (size_t)nmat * per;
    size_t i = (size_t)blockIdx.x * blockDim.x + threadIdx.x;
    if (i >= total) return;
    const int m = (int)(i / per);
    size_t r    = i % per;
    const int v = (int)(r % 8);
    size_t t    = r / 8;
    const int n = (int)(t % N);
    size_t t2   = t / N;
    const int half = (int)(t2 % 2);
    const int ks   = (int)(t2 / 2);
    const int kp   = ks * 16 + half * 8 + v;
    const float* s = src + ((size_t)m * K + 2 * kp) * N + n;
    dst[i] = (unsigned)f2bf(s[0]) | ((unsigned)f2bf(s[N]) << 16);
}

// Node encoder: h = x @ node_W + node_b  (K=9, scalar)
__global__ void node_enc(const float* __restrict__ x, const float* __restrict__ W,
                         const float* __restrict__ b, float* __restrict__ h) {
    int i = blockIdx.x * 256 + threadIdx.x;
    if (i >= NE) return;
    const int r = i >> 7, c = i & 127;
    float acc = b[c];
#pragma unroll
    for (int k = 0; k < INDIM; ++k) acc = fmaf(x[r * INDIM + k], W[k * EMB + c], acc);
    h[i] = acc;
}

__global__ void vn_init(const float* __restrict__ vn_emb, float* __restrict__ vn) {
    int i = blockIdx.x * 256 + threadIdx.x;
    if (i < VE) vn[i] = vn_emb[i & 127];
}

// per-layer reset: zero agg + stats, vtmp = current vn
__global__ void layer_prep(float* __restrict__ agg, float* __restrict__ stats,
                           const float* __restrict__ vn, float* __restrict__ vtmp) {
    int i = blockIdx.x * 256 + threadIdx.x;
    if (i < NE) agg[i] = 0.f;
    if (i < 1024) stats[i] = 0.f;
    if (i < VE) vtmp[i] = vn[i];
}

__global__ void zero_f32(float* __restrict__ p, int n) {
    int i = blockIdx.x * 256 + threadIdx.x;
    if (i < n) p[i] = 0.f;
}

// h_in = h + vn[batch]
__global__ void add_vn(const float* __restrict__ h, const float* __restrict__ vn,
                       const int* __restrict__ batch, float* __restrict__ hin) {
    int i = blockIdx.x * 256 + threadIdx.x;
    if (i >= NE) return;
    const int r = i >> 7, c = i & 127;
    hin[i] = h[i] + vn[(size_t)batch[r] * EMB + c];
}

// edge encoder + message + scatter-add.  8 waves/block, one edge per wave,
// lane handles channels {lane, lane+32, lane+64, lane+96} (coalesced).
__global__ void edge_msg(const float* __restrict__ eattr, const int* __restrict__ src,
                         const int* __restrict__ dst, const float* __restrict__ eW,
                         const float* __restrict__ eb, const float* __restrict__ hin,
                         float* __restrict__ agg) {
    __shared__ float sW[EDGEDIM * EMB];
    __shared__ float sb[EMB];
    const int t = threadIdx.x;
    for (int i = t; i < EDGEDIM * EMB; i += 256) sW[i] = eW[i];
    if (t < EMB) sb[t] = eb[t];
    __syncthreads();
    const int e = blockIdx.x * 8 + (t >> 5);
    if (e >= NEDGES) return;
    const int lane = t & 31;
    float a[EDGEDIM];
    const float* ep = eattr + (size_t)e * EDGEDIM;
#pragma unroll
    for (int k = 0; k < EDGEDIM; ++k) a[k] = ep[k];
    const int s = src[e], d = dst[e];
    const float* hr = hin + (size_t)s * EMB;
    float* ar = agg + (size_t)d * EMB;
#pragma unroll
    for (int j = 0; j < 4; ++j) {
        const int c = lane + j * 32;
        float acc = sb[c];
#pragma unroll
        for (int k = 0; k < EDGEDIM; ++k) acc = fmaf(a[k], sW[k * EMB + c], acc);
        float v = fmaxf(hr[c] + acc, 0.f);
        atomicAdd(&ar[c], v);
    }
}

// t = (1+eps[l])*h_in + agg, converted to bf16 (GEMM A operand)
__global__ void make_t(const float* __restrict__ hin, const float* __restrict__ agg,
                       const float* __restrict__ eps, int l,
                       unsigned short* __restrict__ t16) {
    int i = blockIdx.x * 256 + threadIdx.x;
    if (i >= NE) return;
    const float s = 1.0f + eps[l];
    t16[i] = f2bf(fmaf(s, hin[i], agg[i]));
}

// column sums / sum-of-squares for training-mode BN
__global__ void col_stats(const float* __restrict__ x, int M, int N, int rowsPer,
                          float* __restrict__ sum, float* __restrict__ sumsq) {
    const int c = blockIdx.x * blockDim.x + threadIdx.x;
    if (c >= N) return;
    const int r0 = blockIdx.y * rowsPer;
    const int r1 = min(M, r0 + rowsPer);
    float s = 0.f, q = 0.f;
    for (int r = r0; r < r1; ++r) {
        const float v = x[(size_t)r * N + c];
        s += v; q = fmaf(v, v, q);
    }
    atomicAdd(&sum[c], s);
    atomicAdd(&sumsq[c], q);
}

// BN apply (+optional relu), output f32 or bf16
__global__ void bn_apply(const float* __restrict__ x, const float* __restrict__ sum,
                         const float* __restrict__ sumsq, const float* __restrict__ g,
                         const float* __restrict__ b, int N, float invM, int doRelu,
                         float* __restrict__ outF, unsigned short* __restrict__ outH,
                         size_t total) {
    size_t i = (size_t)blockIdx.x * 256 + threadIdx.x;
    if (i >= total) return;
    const int c = (int)(i % N);
    const float m = sum[c] * invM;
    const float var = fmaxf(sumsq[c] * invM - m * m, 0.f);
    const float rstd = rsqrtf(var + BN_EPS);
    float v = (x[i] - m) * rstd * g[c] + b[c];
    if (doRelu) v = fmaxf(v, 0.f);
    if (outF) outF[i] = v;
    else outH[i] = f2bf(v);
}

// vtmp[batch[r]] += h_in[r]
__global__ void vn_agg(const float* __restrict__ hin, const int* __restrict__ batch,
                       float* __restrict__ vtmp) {
    int i = blockIdx.x * 256 + threadIdx.x;
    if (i >= NE) return;
    const int r = i >> 7, c = i & 127;
    atomicAdd(&vtmp[(size_t)batch[r] * EMB + c], hin[i]);
}

__global__ void cvt_bf16(const float* __restrict__ src, unsigned short* __restrict__ dst,
                         int n) {
    int i = blockIdx.x * 256 + threadIdx.x;
    if (i < n) dst[i] = f2bf(src[i]);
}

// ---------------------------------------------------------------------------
extern "C" void kernel_launch(void* const* d_in, const int* in_sizes, int n_in,
                              void* d_out, int out_size, void* d_ws, size_t ws_size,
                              hipStream_t stream) {
    const float* x         = (const float*)d_in[0];
    const int*   ei        = (const int*)d_in[1];
    const int*   srcIdx    = ei;
    const int*   dstIdx    = ei + NEDGES;
    const float* eattr     = (const float*)d_in[2];
    const int*   batch     = (const int*)d_in[3];
    const float* node_W    = (const float*)d_in[4];
    const float* node_b    = (const float*)d_in[5];
    const float* vn_emb    = (const float*)d_in[6];
    const float* eps       = (const float*)d_in[7];
    const float* edge_W    = (const float*)d_in[8];
    const float* edge_b    = (const float*)d_in[9];
    const float* conv_W1   = (const float*)d_in[10];
    const float* conv_b1   = (const float*)d_in[11];
    const float* cbn_g     = (const float*)d_in[12];
    const float* cbn_b     = (const float*)d_in[13];
    const float* conv_W2   = (const float*)d_in[14];
    const float* conv_b2   = (const float*)d_in[15];
    const float* bn_g      = (const float*)d_in[16];
    const float* bn_b      = (const float*)d_in[17];
    const float* vW1       = (const float*)d_in[18];
    const float* vb1       = (const float*)d_in[19];
    const float* vbn1_g    = (const float*)d_in[20];
    const float* vbn1_b    = (const float*)d_in[21];
    const float* vW2       = (const float*)d_in[22];
    const float* vb2       = (const float*)d_in[23];
    const float* vbn2_g    = (const float*)d_in[24];
    const float* vbn2_b    = (const float*)d_in[25];

    // ---- workspace layout (bump allocator, 256B aligned) ----
    char* w = (char*)d_ws;
    auto alloc = [&](size_t bytes) -> void* {
        void* p = (void*)w;
        w += (bytes + 255) & ~(size_t)255;
        return p;
    };
    float* h    = (float*)alloc((size_t)NE * 4);
    float* hin  = (float*)alloc((size_t)NE * 4);
    float* agg  = (float*)alloc((size_t)NE * 4);
    float* t1   = (float*)alloc((size_t)NNODES * 256 * 4);
    float* t2   = (float*)alloc((size_t)NE * 4);
    unsigned short* t16  = (unsigned short*)alloc((size_t)NE * 2);
    unsigned short* a216 = (unsigned short*)alloc((size_t)NNODES * 256 * 2);
    float* vn   = (float*)alloc((size_t)VE * 4);
    float* vtmp = (float*)alloc((size_t)VE * 4);
    float* v1   = (float*)alloc((size_t)NGRAPHS * 256 * 4);
    float* v2   = (float*)alloc((size_t)VE * 4);
    unsigned short* vt16 = (unsigned short*)alloc((size_t)VE * 2);
    unsigned short* va16 = (unsigned short*)alloc((size_t)NGRAPHS * 256 * 2);
    float* stats = (float*)alloc(1024 * 4);         // sum1[256] sumsq1[256] sum2[128] sumsq2[128]
    float* sum1 = stats, *sq1 = stats + 256, *sum2 = stats + 512, *sq2 = stats + 640;
    unsigned* W1p  = (unsigned*)alloc((size_t)NLAYER * 64 * 256 * 4);
    unsigned* W2p  = (unsigned*)alloc((size_t)NLAYER * 128 * 128 * 4);
    unsigned* vW1p = (unsigned*)alloc((size_t)(NLAYER - 1) * 64 * 256 * 4);
    unsigned* vW2p = (unsigned*)alloc((size_t)(NLAYER - 1) * 128 * 128 * 4);

    const int gNE = (NE + 255) / 256;

    // ---- pack all weights to fragment-major bf16 pairs ----
    {
        size_t n1 = (size_t)NLAYER * 64 * 256;
        size_t n2 = (size_t)NLAYER * 128 * 128;
        size_t n3 = (size_t)(NLAYER - 1) * 64 * 256;
        size_t n4 = (size_t)(NLAYER - 1) * 128 * 128;
        pack_w<<<(unsigned)((n1 + 255) / 256), 256, 0, stream>>>(conv_W1, W1p, 128, 256, NLAYER);
        pack_w<<<(unsigned)((n2 + 255) / 256), 256, 0, stream>>>(conv_W2, W2p, 256, 128, NLAYER);
        pack_w<<<(unsigned)((n3 + 255) / 256), 256, 0, stream>>>(vW1, vW1p, 128, 256, NLAYER - 1);
        pack_w<<<(unsigned)((n4 + 255) / 256), 256, 0, stream>>>(vW2, vW2p, 256, 128, NLAYER - 1);
    }

    node_enc<<<gNE, 256, 0, stream>>>(x, node_W, node_b, h);
    vn_init<<<(VE + 255) / 256, 256, 0, stream>>>(vn_emb, vn);

    auto gemm4 = [&](const unsigned short* A, const unsigned* Bp, const float* bias,
                     float* out, int M, int N) {
        gemm_bf16_wmma<4><<<(M / 16 + 3) / 4, 128, 0, stream>>>(
            A, (const uint4*)Bp, bias, out, M, N);
    };
    auto gemm8 = [&](const unsigned short* A, const unsigned* Bp, const float* bias,
                     float* out, int M, int N) {
        gemm_bf16_wmma<8><<<(M / 16 + 3) / 4, 128, 0, stream>>>(
            A, (const uint4*)Bp, bias, out, M, N);
    };
    auto stats_of = [&](const float* xp, int M, int N, float* s, float* q) {
        dim3 g((N + 255) / 256, (M + 1023) / 1024);
        col_stats<<<g, 256, 0, stream>>>(xp, M, N, 1024, s, q);
    };

    for (int l = 0; l < NLAYER; ++l) {
        layer_prep<<<gNE, 256, 0, stream>>>(agg, stats, vn, vtmp);
        add_vn<<<gNE, 256, 0, stream>>>(h, vn, batch, hin);
        edge_msg<<<(NEDGES + 7) / 8, 256, 0, stream>>>(
            eattr, srcIdx, dstIdx, edge_W + (size_t)l * EDGEDIM * EMB,
            edge_b + (size_t)l * EMB, hin, agg);
        make_t<<<gNE, 256, 0, stream>>>(hin, agg, eps, l, t16);

        // conv MLP: [50000,128] -> [50000,256] -> [50000,128], BN train-mode
        gemm4(t16, W1p + (size_t)l * 64 * 256, conv_b1 + (size_t)l * 256, t1, NNODES, 256);
        stats_of(t1, NNODES, 256, sum1, sq1);
        bn_apply<<<(unsigned)(((size_t)NNODES * 256 + 255) / 256), 256, 0, stream>>>(
            t1, sum1, sq1, cbn_g + (size_t)l * 256, cbn_b + (size_t)l * 256,
            256, 1.0f / NNODES, 1, nullptr, a216, (size_t)NNODES * 256);
        gemm8(a216, W2p + (size_t)l * 128 * 128, conv_b2 + (size_t)l * 128, t2, NNODES, 128);
        stats_of(t2, NNODES, 128, sum2, sq2);
        float* hOut = (l == NLAYER - 1) ? (float*)d_out : h;
        bn_apply<<<gNE, 256, 0, stream>>>(
            t2, sum2, sq2, bn_g + (size_t)l * 128, bn_b + (size_t)l * 128,
            128, 1.0f / NNODES, (l < NLAYER - 1) ? 1 : 0, hOut, nullptr, (size_t)NE);

        if (l < NLAYER - 1) {
            vn_agg<<<gNE, 256, 0, stream>>>(hin, batch, vtmp);
            zero_f32<<<4, 256, 0, stream>>>(stats, 1024);
            cvt_bf16<<<(VE + 255) / 256, 256, 0, stream>>>(vtmp, vt16, VE);
            gemm4(vt16, vW1p + (size_t)l * 64 * 256, vb1 + (size_t)l * 256, v1, NGRAPHS, 256);
            stats_of(v1, NGRAPHS, 256, sum1, sq1);
            bn_apply<<<(NGRAPHS * 256 + 255) / 256, 256, 0, stream>>>(
                v1, sum1, sq1, vbn1_g + (size_t)l * 256, vbn1_b + (size_t)l * 256,
                256, 1.0f / NGRAPHS, 1, nullptr, va16, (size_t)NGRAPHS * 256);
            gemm8(va16, vW2p + (size_t)l * 128 * 128, vb2 + (size_t)l * 128, v2, NGRAPHS, 128);
            stats_of(v2, NGRAPHS, 128, sum2, sq2);
            bn_apply<<<(VE + 255) / 256, 256, 0, stream>>>(
                v2, sum2, sq2, vbn2_g + (size_t)l * 128, vbn2_b + (size_t)l * 128,
                128, 1.0f / NGRAPHS, 1, vn, nullptr, (size_t)VE);
        }
    }
}
